// DiscriminatorRNN_16827681866111
// MI455X (gfx1250) — compile-verified, matrix operand order checked by reading, not run on hardware
//
#include <hip/hip_runtime.h>
#include <hip/hip_bf16.h>
#include <math.h>

// ---------------------------------------------------------------------------
// DiscriminatorRNN for MI455X (gfx1250, wave32, WMMA bf16 16x16x32, f32 acc)
// ---------------------------------------------------------------------------

typedef __bf16 bf16_t;
typedef __attribute__((ext_vector_type(16))) __bf16 v16bf;
typedef __attribute__((ext_vector_type(8)))  __bf16 v8bf;
typedef __attribute__((ext_vector_type(8)))  float  v8f;

#define BATCH  1024
#define SRC    50
#define NT     74          // total frames T
#define ISZ    69          // input feature size
#define IPAD   96          // padded K for x / w_ih (multiple of 32)
#define HSZ    1024
#define H3     3072
#define NPAD   80          // fc1 N padded to 5 * 16

// float -> bf16, round to nearest even
__device__ __forceinline__ bf16_t f2bf(float f) {
  union { float f; unsigned u; } v; v.f = f;
  unsigned r = v.u + 0x7FFFu + ((v.u >> 16) & 1u);
  unsigned short h = (unsigned short)(r >> 16);
  return __builtin_bit_cast(bf16_t, h);
}

__device__ __forceinline__ float sigmf(float x) { return 1.0f / (1.0f + expf(-x)); }

// Build a 16x32 bf16 WMMA operand fragment from a row-major [rows x ld] bf16
// matrix. Per the CDNA5 ISA 16-bit A/B layout: lane L<16 holds row (row0+L),
// K = {k0..k0+7, k0+16..k0+23}; lane L>=16 holds K = {k0+8.., k0+24..}.
// Both halves are contiguous 16-byte reads -> global_load_b128.
__device__ __forceinline__ v16bf load_frag(const bf16_t* __restrict__ base,
                                           int row0, int k0, int ld) {
  const int lane = __lane_id();
  const int half = lane >> 4;                      // 0 or 1
  const bf16_t* p = base + (size_t)(row0 + (lane & 15)) * ld + k0 + half * 8;
  v8bf lo = *(const v8bf*)(p);                     // K: k0 + half*8 .. +7
  v8bf hi = *(const v8bf*)(p + 16);                // K: k0+16 + half*8 .. +7
  v16bf out;
#pragma unroll
  for (int i = 0; i < 8; ++i) { out[i] = lo[i]; out[i + 8] = hi[i]; }
  return out;
}

__device__ __forceinline__ v8f wmma_bf16(v16bf a, v16bf b, v8f c) {
  // D = A(16x32) * B(32x16) + C, f32 accumulate
  return __builtin_amdgcn_wmma_f32_16x16x32_bf16(
      false, a, false, b, (short)0, c, false, false);
}

// ---------------------------------------------------------------------------
// Prep kernels: convert / pad weights and frames to bf16
// ---------------------------------------------------------------------------

__global__ void k_cvt_whh(const float* __restrict__ w, bf16_t* __restrict__ o) {
  int i = blockIdx.x * blockDim.x + threadIdx.x;        // [3072 * 1024]
  if (i < H3 * HSZ) o[i] = f2bf(w[i]);
}

__global__ void k_cvt_wih(const float* __restrict__ w, bf16_t* __restrict__ o) {
  int i = blockIdx.x * blockDim.x + threadIdx.x;        // [3072 * 96]
  if (i >= H3 * IPAD) return;
  int r = i / IPAD, c = i - r * IPAD;
  o[i] = (c < ISZ) ? f2bf(w[r * ISZ + c]) : f2bf(0.0f);
}

__global__ void k_cvt_fc1(const float* __restrict__ w, bf16_t* __restrict__ o) {
  int i = blockIdx.x * blockDim.x + threadIdx.x;        // [80 * 1024]
  if (i >= NPAD * HSZ) return;
  int r = i / HSZ, c = i - r * HSZ;
  o[i] = (r < ISZ) ? f2bf(w[r * HSZ + c]) : f2bf(0.0f);
}

// frames[t][b][c] padded to IPAD, from concat(encoder_inputs, decoder_output)
__global__ void k_frames(const float* __restrict__ enc, const float* __restrict__ dec,
                         bf16_t* __restrict__ o) {
  int i = blockIdx.x * blockDim.x + threadIdx.x;        // [74 * 1024 * 96]
  if (i >= NT * BATCH * IPAD) return;
  int c = i % IPAD;
  int tb = i / IPAD;
  int b = tb % BATCH;
  int t = tb / BATCH;
  float v = 0.0f;
  if (c < ISZ)
    v = (t < SRC) ? enc[((size_t)b * SRC + t) * ISZ + c]
                  : dec[((size_t)b * (NT - SRC) + (t - SRC)) * ISZ + c];
  o[i] = f2bf(v);
}

__global__ void k_zero_state(float* __restrict__ hf, bf16_t* __restrict__ hb,
                             bf16_t* __restrict__ xdec) {
  int i = blockIdx.x * blockDim.x + threadIdx.x;
  if (i < HSZ * BATCH) { hf[i] = 0.0f; hb[i] = f2bf(0.0f); }
  if (i < BATCH * IPAD) xdec[i] = f2bf(0.0f);
}

// ---------------------------------------------------------------------------
// Fused GRU step:  h_new = (1-z)*tanh(i_n + r*h_n) + z*h
// One wave owns a 32(batch) x 16(hidden) output tile (two 16x16 M-tiles) so
// each per-gate B fragment is reused twice: 5 fragment loads -> 6 WMMAs.
// Accumulators per M-tile: acc_r, acc_z fused over x@w_ih + h@w_hh;
//                          acc_in (x only); acc_hn (h only).
// Block = 256 threads = 8 waves arranged 2(m) x 4(n): tile 64 x 64.
// Grid (16, 16) = 256 blocks.
// ---------------------------------------------------------------------------
__global__ void __launch_bounds__(256)
k_gru_step(const bf16_t* __restrict__ x,      // [1024][96] bf16, zero-padded
           const bf16_t* __restrict__ h_bf,   // [1024][1024] bf16
           const float*  __restrict__ h_f32,  // [1024][1024] f32 (prev state)
           const bf16_t* __restrict__ w_ih,   // [3072][96]
           const bf16_t* __restrict__ w_hh,   // [3072][1024]
           const float*  __restrict__ b_ih, const float* __restrict__ b_hh,
           float* __restrict__ h_out_f32, bf16_t* __restrict__ h_out_bf) {
  const int lane = __lane_id();
  const int wave = threadIdx.x >> 5;
  const int m0 = blockIdx.x * 64 + (wave & 1) * 32;   // batch tile origin (2 tiles)
  const int n0 = blockIdx.y * 64 + (wave >> 1) * 16;  // hidden tile origin

  v8f accR0 = {}, accZ0 = {}, accI0 = {}, accH0 = {};   // rows m0 .. m0+15
  v8f accR1 = {}, accZ1 = {}, accI1 = {}, accH1 = {};   // rows m0+16 .. m0+31

  // gi contribution: x @ w_ih.T  (K = 96, zero padded)
#pragma unroll
  for (int k0 = 0; k0 < IPAD; k0 += 32) {
    v16bf a0 = load_frag(x, m0,      k0, IPAD);
    v16bf a1 = load_frag(x, m0 + 16, k0, IPAD);
    v16bf br = load_frag(w_ih, n0,           k0, IPAD);
    accR0 = wmma_bf16(a0, br, accR0);
    accR1 = wmma_bf16(a1, br, accR1);
    v16bf bz = load_frag(w_ih, HSZ + n0,     k0, IPAD);
    accZ0 = wmma_bf16(a0, bz, accZ0);
    accZ1 = wmma_bf16(a1, bz, accZ1);
    v16bf bn = load_frag(w_ih, 2 * HSZ + n0, k0, IPAD);
    accI0 = wmma_bf16(a0, bn, accI0);
    accI1 = wmma_bf16(a1, bn, accI1);
  }

  // gh contribution: h @ w_hh.T  (K = 1024)
  for (int k0 = 0; k0 < HSZ; k0 += 32) {
    if (k0 + 32 < HSZ) {
      __builtin_prefetch(h_bf + (size_t)m0 * HSZ + k0 + 32, 0, 1);
      __builtin_prefetch(h_bf + (size_t)(m0 + 16) * HSZ + k0 + 32, 0, 1);
    }
    v16bf a0 = load_frag(h_bf, m0,      k0, HSZ);
    v16bf a1 = load_frag(h_bf, m0 + 16, k0, HSZ);
    v16bf br = load_frag(w_hh, n0,           k0, HSZ);
    accR0 = wmma_bf16(a0, br, accR0);
    accR1 = wmma_bf16(a1, br, accR1);
    v16bf bz = load_frag(w_hh, HSZ + n0,     k0, HSZ);
    accZ0 = wmma_bf16(a0, bz, accZ0);
    accZ1 = wmma_bf16(a1, bz, accZ1);
    v16bf bn = load_frag(w_hh, 2 * HSZ + n0, k0, HSZ);
    accH0 = wmma_bf16(a0, bn, accH0);
    accH1 = wmma_bf16(a1, bn, accH1);
  }

  // Gate epilogue. C/D layout: lane<16 -> M=tile+i, lane>=16 -> M=tile+8+i;
  // N = n0 + (lane & 15).
  const int ncol = n0 + (lane & 15);
  const float br_ = b_ih[ncol]           + b_hh[ncol];
  const float bz_ = b_ih[HSZ + ncol]     + b_hh[HSZ + ncol];
  const float bin = b_ih[2 * HSZ + ncol];
  const float bhn = b_hh[2 * HSZ + ncol];
  const int half8 = (lane >> 4) << 3;

#pragma unroll
  for (int mi = 0; mi < 2; ++mi) {
    const v8f& aR = mi ? accR1 : accR0;
    const v8f& aZ = mi ? accZ1 : accZ0;
    const v8f& aI = mi ? accI1 : accI0;
    const v8f& aH = mi ? accH1 : accH0;
    const int rbase = m0 + mi * 16 + half8;
#pragma unroll
    for (int i = 0; i < 8; ++i) {
      const int brow = rbase + i;
      float r  = sigmf(aR[i] + br_);
      float z  = sigmf(aZ[i] + bz_);
      float nn = tanhf(aI[i] + bin + r * (aH[i] + bhn));
      float hp = h_f32[(size_t)brow * HSZ + ncol];
      float hn = (1.0f - z) * nn + z * hp;
      h_out_f32[(size_t)brow * HSZ + ncol] = hn;
      h_out_bf [(size_t)brow * HSZ + ncol] = f2bf(hn);
    }
  }
}

// ---------------------------------------------------------------------------
// Decode output: out = prev + h @ fc1_w.T + fc1_b  (N = 69 padded to 80)
// Small L2-hot GEMM: maximize parallelism — one 16x16 tile per wave.
// Grid (8, 5): blockIdx.y = N tile; 8 waves/block cover 128 batch rows.
// Writes f32 output frame and bf16 padded next GRU input.
// ---------------------------------------------------------------------------
__global__ void __launch_bounds__(256)
k_fc1_step(const bf16_t* __restrict__ h_bf,    // [1024][1024]
           const float*  __restrict__ prev, int prev_stride,
           const bf16_t* __restrict__ fc1w,    // [80][1024], rows >=69 zero
           const float*  __restrict__ fc1b,
           float* __restrict__ out_t,          // [1024][69]
           bf16_t* __restrict__ x_next) {      // [1024][96], pad cols stay 0
  const int lane = __lane_id();
  const int wave = threadIdx.x >> 5;
  const int m0 = blockIdx.x * 128 + wave * 16;
  const int nt = blockIdx.y;                   // 0..4

  v8f acc = {};
  for (int k0 = 0; k0 < HSZ; k0 += 32) {
    v16bf a = load_frag(h_bf, m0, k0, HSZ);
    v16bf b = load_frag(fc1w, nt * 16, k0, HSZ);
    acc = wmma_bf16(a, b, acc);
  }

  const int ncol = nt * 16 + (lane & 15);
  if (ncol < ISZ) {
    const float bias = fc1b[ncol];
    const int rbase = m0 + ((lane >> 4) << 3);
#pragma unroll
    for (int i = 0; i < 8; ++i) {
      const int brow = rbase + i;
      float v = prev[(size_t)brow * prev_stride + ncol] + acc[i] + bias;
      out_t [(size_t)brow * ISZ  + ncol] = v;
      x_next[(size_t)brow * IPAD + ncol] = f2bf(v);
    }
  }
}

// ---------------------------------------------------------------------------
// Final: logit[b] = sigmoid( sum_{t,i} out[t][b][i] * fc2_w[t*69+i] + fc2_b )
// ---------------------------------------------------------------------------
__global__ void __launch_bounds__(256)
k_final(const float* __restrict__ outs,       // [74][1024][69]
        const float* __restrict__ fc2w, const float* __restrict__ fc2b,
        float* __restrict__ out) {
  const int b = blockIdx.x, tid = threadIdx.x;
  float s = 0.0f;
  for (int j = tid; j < NT * ISZ; j += 256) {
    int t = j / ISZ, i = j - t * ISZ;
    s += outs[((size_t)t * BATCH + b) * ISZ + i] * fc2w[j];
  }
  __shared__ float red[256];
  red[tid] = s;
  __syncthreads();
  for (int off = 128; off > 0; off >>= 1) {
    if (tid < off) red[tid] += red[tid + off];
    __syncthreads();
  }
  if (tid == 0) out[b] = sigmf(red[0] + fc2b[0]);
}

// ---------------------------------------------------------------------------
// Host orchestration
// ---------------------------------------------------------------------------
static inline size_t align256(size_t x) { return (x + 255) & ~(size_t)255; }

extern "C" void kernel_launch(void* const* d_in, const int* in_sizes, int n_in,
                              void* d_out, int out_size, void* d_ws, size_t ws_size,
                              hipStream_t stream) {
  const float* enc   = (const float*)d_in[0];   // [1024,50,69]
  const float* dec   = (const float*)d_in[1];   // [1024,24,69]
  const float* w_ih  = (const float*)d_in[2];   // [3072,69]
  const float* w_hh  = (const float*)d_in[3];   // [3072,1024]
  const float* b_ih  = (const float*)d_in[4];   // [3072]
  const float* b_hh  = (const float*)d_in[5];   // [3072]
  const float* fc1_w = (const float*)d_in[6];   // [69,1024]
  const float* fc1_b = (const float*)d_in[7];   // [69]
  const float* fc2_w = (const float*)d_in[8];   // [1,74*69]
  const float* fc2_b = (const float*)d_in[9];   // [1]
  float* out = (float*)d_out;                   // [1024,1]

  char* ws = (char*)d_ws;
  size_t off = 0;
  bf16_t* whh_bf  = (bf16_t*)(ws + off); off = align256(off + (size_t)H3 * HSZ * 2);
  bf16_t* wih_bf  = (bf16_t*)(ws + off); off = align256(off + (size_t)H3 * IPAD * 2);
  bf16_t* fc1w_bf = (bf16_t*)(ws + off); off = align256(off + (size_t)NPAD * HSZ * 2);
  bf16_t* x_bf    = (bf16_t*)(ws + off); off = align256(off + (size_t)NT * BATCH * IPAD * 2);
  float*  hfA     = (float*) (ws + off); off = align256(off + (size_t)BATCH * HSZ * 4);
  float*  hfB     = (float*) (ws + off); off = align256(off + (size_t)BATCH * HSZ * 4);
  bf16_t* hbA     = (bf16_t*)(ws + off); off = align256(off + (size_t)BATCH * HSZ * 2);
  bf16_t* hbB     = (bf16_t*)(ws + off); off = align256(off + (size_t)BATCH * HSZ * 2);
  float*  outs    = (float*) (ws + off); off = align256(off + (size_t)NT * BATCH * ISZ * 4);
  bf16_t* x_dec   = (bf16_t*)(ws + off); off = align256(off + (size_t)BATCH * IPAD * 2);
  (void)ws_size; (void)n_in; (void)in_sizes; (void)out_size;

  // --- prep ---
  k_cvt_whh <<<(H3 * HSZ + 255) / 256, 256, 0, stream>>>(w_hh, whh_bf);
  k_cvt_wih <<<(H3 * IPAD + 255) / 256, 256, 0, stream>>>(w_ih, wih_bf);
  k_cvt_fc1 <<<(NPAD * HSZ + 255) / 256, 256, 0, stream>>>(fc1_w, fc1w_bf);
  k_frames  <<<(NT * BATCH * IPAD + 255) / 256, 256, 0, stream>>>(enc, dec, x_bf);
  k_zero_state<<<(BATCH * HSZ + 255) / 256, 256, 0, stream>>>(hfA, hbA, x_dec);

  float*  hf[2] = {hfA, hfB};
  bf16_t* hb[2] = {hbA, hbB};
  int cur = 0;
  const dim3 gru_grid(BATCH / 64, HSZ / 64);   // (16, 16)

  // --- encode: 74 GRU steps over all frames ---
  for (int t = 0; t < NT; ++t) {
    k_gru_step<<<gru_grid, 256, 0, stream>>>(
        x_bf + (size_t)t * BATCH * IPAD, hb[cur], hf[cur],
        wih_bf, whh_bf, b_ih, b_hh, hf[1 - cur], hb[1 - cur]);
    cur ^= 1;
  }

  // --- decode: 74 autoregressive (GRU + fc1-residual) steps ---
  for (int d = 0; d < NT; ++d) {
    const bf16_t* xin = (d == 0) ? x_bf : x_dec;   // frame 0, then prev output
    k_gru_step<<<gru_grid, 256, 0, stream>>>(
        xin, hb[cur], hf[cur], wih_bf, whh_bf, b_ih, b_hh,
        hf[1 - cur], hb[1 - cur]);
    cur ^= 1;
    const float* prev = (d == 0) ? enc : (outs + (size_t)(d - 1) * BATCH * ISZ);
    const int pstride = (d == 0) ? SRC * ISZ : ISZ;  // frames[0] = enc[:,0,:]
    k_fc1_step<<<dim3(BATCH / 128, 5), 256, 0, stream>>>(
        hb[cur], prev, pstride, fc1w_bf, fc1_b,
        outs + (size_t)d * BATCH * ISZ, x_dec);
  }

  // --- final logit ---
  k_final<<<BATCH, 256, 0, stream>>>(outs, fc2_w, fc2_b, out);
}
